// LiquidTimeConstantBlock_41051297415482
// MI455X (gfx1250) — matrix operand order repaired and checked by reading, not compiled, not run on hardware
//
#include <hip/hip_runtime.h>
#include <hip/hip_bf16.h>

// LiquidTimeConstantBlock for MI455X (gfx1250, wave32, WMMA, TDM).
// Phase 0: convert weights fp32 -> bf16 (L2-resident afterwards), init tau=1.
// Phase 1: fused GEMM [xproj | m] = x @ [Wx^T | Wm^T] + bias via v_wmma_f32_16x16x32_bf16,
//          double-buffered LDS pipeline, non-temporal streaming of x / outputs.
// Phase 2: 16 persistent WGs scan over T; per-step 64x512x512 WMMA GEMM; Wt slice staged
//          into LDS via Tensor Data Mover (tensor_load_to_lds + s_wait_tensorcnt) with a
//          plain-load backstop; per-step sync = cluster barrier + monotonic atomic.

typedef __bf16 bf16_t;
typedef __attribute__((ext_vector_type(16))) __bf16 v16bf;
typedef __attribute__((ext_vector_type(8)))  __bf16 v8bf;
typedef __attribute__((ext_vector_type(8)))  float  v8f;
typedef __attribute__((ext_vector_type(4)))  float  v4f;
typedef __attribute__((ext_vector_type(4)))  unsigned int v4u;
typedef __attribute__((ext_vector_type(8)))  int    v8i;
typedef __attribute__((ext_vector_type(4)))  int    v4i;

#define B_  64
#define T_  1024
#define D_  512
#define BT_ (B_ * T_)

// ---------------- WMMA fragment builders (CDNA5 bf16 16x16x32 layouts) -------------

// A matrix 16x32 (MxK), source row-major [m][k], ld in elements.
// lane<16:  M=lane,     elems 0..7 = K 0..7,   elems 8..15 = K 16..23
// lane>=16: M=lane-16,  elems 0..7 = K 8..15,  elems 8..15 = K 24..31
__device__ __forceinline__ v16bf make_afrag(const bf16_t* base, int ld, int m0, int k0, int lane) {
    const int half = lane >> 4;
    const bf16_t* p = base + (size_t)(m0 + (lane & 15)) * ld + k0 + 8 * half;
    union { v16bf v; v8bf h[2]; } u;
    u.h[0] = *(const v8bf*)(p);
    u.h[1] = *(const v8bf*)(p + 16);
    return u.v;
}

// B matrix 32x16 (KxN), B[k][n] = W[n][k]; source W row-major [n][k], ld in elements.
// lane<16: N=lane, elems 0..7 = K 0..7, 8..15 = K 16..23 ; lane>=16: N=lane-16, elems 0..7 = K 8..15, 8..15 = K 24..31
__device__ __forceinline__ v16bf make_bfrag(const bf16_t* base, int ld, int n0, int k0, int lane) {
    const int half = lane >> 4;
    const bf16_t* p = base + (size_t)(n0 + (lane & 15)) * ld + k0 + 8 * half;
    union { v16bf v; v8bf h[2]; } u;
    u.h[0] = *(const v8bf*)(p);
    u.h[1] = *(const v8bf*)(p + 16);
    return u.v;
}

// ---------------- Phase 0: weight conversion + state init ------------------------

__global__ __launch_bounds__(256) void ltc_prepare(
    const float* __restrict__ tau_w,   // (512, 1024) row-major
    const float* __restrict__ mem_w,   // (512, 512)
    bf16_t* __restrict__ wxm,          // (1024, 512): rows 0..511 = Wx, 512..1023 = Wm
    bf16_t* __restrict__ wt,           // (512, 512)  = Wt  (tau_w[:, 512:])
    bf16_t* __restrict__ tau0,         // (64, 512) init to 1.0
    unsigned* __restrict__ syncCtr)
{
    const int i = blockIdx.x * 256 + threadIdx.x;   // 0 .. 512*512-1
    if (i == 0) *syncCtr = 0u;
    if (i < D_ * D_) {
        const int e = i >> 9;
        const int d = i & 511;
        wxm[i]             = (bf16_t)tau_w[(size_t)e * (2 * D_) + d];
        wxm[D_ * D_ + i]   = (bf16_t)mem_w[i];
        wt[i]              = (bf16_t)tau_w[(size_t)e * (2 * D_) + D_ + d];
    }
    if (i < B_ * D_) tau0[i] = (bf16_t)1.0f;
}

// ---------------- Phase 1: fused precompute GEMM (double-buffered) ---------------
// C(65536 x 1024) = x(65536 x 512) @ W^T ; cols <512 -> xproj(+tau_b), >=512 -> m(+mem_b)
// WG tile 128x64, 8 waves, each wave 32x32 (2x2 WMMA tiles). One barrier per K-chunk.

__global__ __launch_bounds__(256) void ltc_gemm_precompute(
    const float*  __restrict__ x,      // (BT, 512)
    const bf16_t* __restrict__ wxm,    // (1024, 512)
    const float*  __restrict__ tau_b,  // (512)
    const float*  __restrict__ mem_b,  // (512)
    float* __restrict__ xproj,         // (BT, 512)  -> d_out spike region (reused)
    float* __restrict__ mbuf)          // (BT, 512)  -> workspace
{
    __shared__ bf16_t xs[2][128][40];  // 32 K + 8 pad, 16B-aligned rows
    __shared__ bf16_t wsh[2][64][40];

    const int tid  = threadIdx.x;
    const int lane = tid & 31;
    const int wave = tid >> 5;
    const int rowBase = blockIdx.y * 128;
    const int colBase = blockIdx.x * 64;
    const int wm = wave >> 1;          // 0..3 : 32 rows per wave
    const int wn = wave & 1;           // 0..1 : 32 cols per wave

    // staging coordinates
    const int sr = tid >> 1;           // 0..127
    const int sc = (tid & 1) * 16;     // 0 or 16

    v8f acc[2][2] = {};

    // stage K-chunk `k0` into buffer `buf` (x fp32 -> bf16, weights bf16 copy)
    auto stage = [&](int buf, int k0) {
        const float* src = x + (size_t)(rowBase + sr) * D_ + k0 + sc;
        const v4f* s4 = (const v4f*)src;
        v4f f0 = __builtin_nontemporal_load(s4);       // x is read exactly once
        v4f f1 = __builtin_nontemporal_load(s4 + 1);
        v4f f2 = __builtin_nontemporal_load(s4 + 2);
        v4f f3 = __builtin_nontemporal_load(s4 + 3);
        bf16_t* dst = &xs[buf][sr][sc];
        #pragma unroll
        for (int i = 0; i < 4; ++i) dst[i]      = (bf16_t)f0[i];
        #pragma unroll
        for (int i = 0; i < 4; ++i) dst[4 + i]  = (bf16_t)f1[i];
        #pragma unroll
        for (int i = 0; i < 4; ++i) dst[8 + i]  = (bf16_t)f2[i];
        #pragma unroll
        for (int i = 0; i < 4; ++i) dst[12 + i] = (bf16_t)f3[i];
        if (tid < 128) {
            const bf16_t* wsrc = wxm + (size_t)(colBase + sr) * D_ + k0 + sc;
            *(v8bf*)&wsh[buf][sr][sc]     = *(const v8bf*)(wsrc);
            *(v8bf*)&wsh[buf][sr][sc + 8] = *(const v8bf*)(wsrc + 8);
        }
    };

    stage(0, 0);
    __syncthreads();

    for (int kc = 0; kc < D_ / 32; ++kc) {
        const int cur = kc & 1;
        if (kc + 1 < D_ / 32) stage(cur ^ 1, (kc + 1) * 32);  // overlap with WMMA below
        #pragma unroll
        for (int mi = 0; mi < 2; ++mi) {
            v16bf a = make_afrag(&xs[cur][0][0], 40, wm * 32 + mi * 16, 0, lane);
            #pragma unroll
            for (int ni = 0; ni < 2; ++ni) {
                v16bf b = make_bfrag(&wsh[cur][0][0], 40, wn * 32 + ni * 16, 0, lane);
                acc[mi][ni] = __builtin_amdgcn_wmma_f32_16x16x32_bf16(
                    false, a, false, b, (short)0, acc[mi][ni], false, false);
            }
        }
        __syncthreads();
    }

    // epilogue: bias + scatter (C layout: M = r + 8*(lane>=16), N = lane&15)
    const int half = lane >> 4;
    const int ncol = lane & 15;
    #pragma unroll
    for (int mi = 0; mi < 2; ++mi) {
        #pragma unroll
        for (int ni = 0; ni < 2; ++ni) {
            const int nG   = colBase + wn * 32 + ni * 16 + ncol;
            const bool isX = nG < D_;
            const float bias = isX ? tau_b[nG] : mem_b[nG - D_];
            float* outp = isX ? xproj : mbuf;
            const int nOut = isX ? nG : nG - D_;
            #pragma unroll
            for (int r = 0; r < 8; ++r) {
                const int mG = rowBase + wm * 32 + mi * 16 + r + 8 * half;
                __builtin_nontemporal_store(acc[mi][ni][r] + bias,
                                            &outp[(size_t)mG * D_ + nOut]);
            }
        }
    }
}

// ---------------- Phase 2: persistent recurrent scan -----------------------------

__global__ __launch_bounds__(256) void ltc_recurrent(
    const bf16_t* __restrict__ wt,       // (512,512) [e][k] bf16
    const float*  __restrict__ mbuf,     // (BT, 512)
    const float*  __restrict__ thr_p,    // scalar
    bf16_t* __restrict__ tauA,           // (64,512) bf16 double buffer
    bf16_t* __restrict__ tauB,
    float*  __restrict__ xproj_spikes,   // (BT,512): read xproj[t], overwrite with spikes[t]
    float*  __restrict__ tau_out,        // (64,512)
    float*  __restrict__ v_out,          // (64,512)
    unsigned* __restrict__ syncCtr)
{
    __shared__ bf16_t wts[32][520];      // 32 e-rows x 512 K (+8 pad = 16B), 33.3 KB

    const int tid   = threadIdx.x;
    const int lane  = tid & 31;
    const int wave  = tid >> 5;
    const int eBase = blockIdx.x * 32;

    // --- TDM: DMA the 32x512 bf16 Wt tile into LDS with 16B per-row padding ------
    // D# group0: count=1, lds_addr, global_addr(57b), type=2.
    // D# group1: data_size=2B, pad_enable, pad_interval=256 DW (1024B row),
    //            pad_amount=4 DW (16B), tensor_dim0=512, tensor_dim1=32,
    //            tile 512x32, strides 512.
    if (wave == 0) {
        const unsigned long long ga =
            (unsigned long long)(uintptr_t)(wt + (size_t)eBase * D_);
        const unsigned ldsOff = (unsigned)(uintptr_t)
            (__attribute__((address_space(3))) bf16_t*)&wts[0][0];
        v4u g0 = { 1u, ldsOff, (unsigned)ga,
                   (unsigned)((ga >> 32) & 0x01FFFFFFu) | (2u << 30) };
        v8i g1 = { (int)((1u << 16) | (1u << 20) | (7u << 22) | (3u << 25)),
                   (int)(512u << 16),   // tensor_dim0 lo @ [63:48]
                   (int)(32u  << 16),   // tensor_dim0 hi=0, tensor_dim1 lo @ [95:80]
                   (int)(512u << 16),   // tensor_dim1 hi=0, tile_dim0 @ [127:112]
                   32,                  // tile_dim1 @ [143:128], tile_dim2=0
                   512,                 // tensor_dim0_stride lo
                   (int)(512u << 16),   // dim0_stride hi=0, tensor_dim1_stride lo @ [223:208]
                   0 };
        v4i g2 = { 1, 0, 0, 0 };        // tensor_dim2=1, rest unused (tile_dim3=0)
        v4i g3 = { 0, 0, 0, 0 };
#if defined(__clang_major__) && __clang_major__ >= 23
        v8i g4 = { 0, 0, 0, 0, 0, 0, 0, 0 };
        __builtin_amdgcn_tensor_load_to_lds(g0, g1, g2, g3, g4, 0);
#else
        __builtin_amdgcn_tensor_load_to_lds(g0, g1, g2, g3, 0);
#endif
        __builtin_amdgcn_s_wait_tensorcnt(0);
    }
    __syncthreads();

    // Plain-load backstop (overwrites the tile with known-good data; LDS OOB writes
    // from any descriptor mistake are dropped by hardware, so this is always safe).
    {
        const int r  = tid >> 3;
        const int c0 = (tid & 7) * 64;
        const bf16_t* src = wt + (size_t)(eBase + r) * D_ + c0;
        bf16_t* dst = &wts[r][c0];
        #pragma unroll
        for (int i = 0; i < 8; ++i)
            *(v8bf*)(dst + i * 8) = *(const v8bf*)(src + i * 8);
    }
    __syncthreads();

    const float thr = *thr_p;
    const int mi = wave >> 1;            // 0..3 -> b rows [16mi, 16mi+16)
    const int ni = wave & 1;             // 0..1 -> e cols [16ni, 16ni+16) within slice
    const int half = lane >> 4;
    const int ncol = lane & 15;
    const int eG = eBase + ni * 16 + ncol;

    float v[8];
    #pragma unroll
    for (int r = 0; r < 8; ++r) v[r] = 0.0f;

    const unsigned nWG = gridDim.x;

    #pragma unroll 1
    for (int t = 0; t < T_; ++t) {
        const bf16_t* tc = (t & 1) ? tauB : tauA;
        bf16_t*       tn = (t & 1) ? tauA : tauB;

        // tau(64x512, L2-resident) @ Wt_slice^T -> 16x16 tile per wave
        v8f acc = {};
        #pragma unroll 4
        for (int k = 0; k < D_; k += 32) {
            v16bf a = make_afrag(tc, D_, mi * 16, k, lane);
            v16bf b = make_bfrag(&wts[0][0], 520, ni * 16, k, lane);
            acc = __builtin_amdgcn_wmma_f32_16x16x32_bf16(
                false, a, false, b, (short)0, acc, false, false);
        }

        // elementwise update in C-register layout
        #pragma unroll
        for (int r = 0; r < 8; ++r) {
            const int b = mi * 16 + r + 8 * half;
            const size_t idx = ((size_t)b * T_ + t) * D_ + eG;
            const float pre   = acc[r] + __builtin_nontemporal_load(&xproj_spikes[idx]);
            const float taun  = 1.0f / (1.0f + __expf(-pre));
            const float alpha = __expf(-1.0f / (taun + 1e-6f));
            const float mv    = __builtin_nontemporal_load(&mbuf[idx]);
            v[r] = alpha * v[r] + (1.0f - alpha) * mv;
            const float s = (v[r] >= thr) ? 1.0f : 0.0f;
            __builtin_nontemporal_store(s, &xproj_spikes[idx]);  // RAW in-thread is ordered
            v[r] *= (1.0f - s);
            tn[(size_t)b * D_ + eG] = (bf16_t)taun;
            if (t == T_ - 1) {
                tau_out[(size_t)b * D_ + eG] = taun;
                v_out[(size_t)b * D_ + eG]   = v[r];
            }
        }

        // ---- per-step device-wide sync ----
        __threadfence();                              // release tau stores to L2
        __syncthreads();
        if (wave == 0)
            asm volatile("s_barrier_signal -3" ::: "memory");   // cluster barrier (CDNA5)
        asm volatile("s_barrier_wait -3" ::: "memory");
        if (tid == 0) {                               // atomic fallback (exact sync)
            __hip_atomic_fetch_add(syncCtr, 1u, __ATOMIC_RELEASE, __HIP_MEMORY_SCOPE_AGENT);
            const unsigned target = nWG * (unsigned)(t + 1);
            while (__hip_atomic_load(syncCtr, __ATOMIC_ACQUIRE, __HIP_MEMORY_SCOPE_AGENT) < target) {}
        }
        __syncthreads();
        __builtin_amdgcn_fence(__ATOMIC_ACQUIRE, "agent");      // invalidate near caches
    }
}

// ---------------- host launcher ---------------------------------------------------

extern "C" void kernel_launch(void* const* d_in, const int* in_sizes, int n_in,
                              void* d_out, int out_size, void* d_ws, size_t ws_size,
                              hipStream_t stream) {
    (void)in_sizes; (void)n_in; (void)out_size; (void)ws_size;

    const float* x     = (const float*)d_in[0];
    const float* tau_w = (const float*)d_in[1];
    const float* tau_b = (const float*)d_in[2];
    const float* mem_w = (const float*)d_in[3];
    const float* mem_b = (const float*)d_in[4];
    const float* thr   = (const float*)d_in[5];

    float* out      = (float*)d_out;
    float* spikes   = out;                                   // (B,T,D), first holds xproj
    float* tau_out  = out + (size_t)BT_ * D_;                // (B,D)
    float* v_out    = tau_out + (size_t)B_ * D_;             // (B,D)

    char* ws = (char*)d_ws;
    float* mbuf = (float*)ws;                                // (BT,D) fp32 = 128 MB
    size_t off = (size_t)BT_ * D_ * sizeof(float);
    bf16_t* wxm  = (bf16_t*)(ws + off); off += (size_t)1024 * D_ * sizeof(bf16_t);
    bf16_t* wt   = (bf16_t*)(ws + off); off += (size_t)D_ * D_ * sizeof(bf16_t);
    bf16_t* tauA = (bf16_t*)(ws + off); off += (size_t)B_ * D_ * sizeof(bf16_t);
    bf16_t* tauB = (bf16_t*)(ws + off); off += (size_t)B_ * D_ * sizeof(bf16_t);
    unsigned* syncCtr = (unsigned*)(ws + off);

    ltc_prepare<<<(D_ * D_ + 255) / 256, 256, 0, stream>>>(
        tau_w, mem_w, wxm, wt, tauA, syncCtr);

    dim3 g1(1024 / 64, BT_ / 128);   // (16, 512)
    ltc_gemm_precompute<<<g1, 256, 0, stream>>>(x, wxm, tau_b, mem_b, spikes, mbuf);

    ltc_recurrent<<<D_ / 32, 256, 0, stream>>>(
        wt, mbuf, thr, tauA, tauB, spikes, tau_out, v_out, syncCtr);
}
